// TorchBatchMacenkoNormalizer_61194694033554
// MI455X (gfx1250) — compile-verified, matrix operand order checked by reading, not run on hardware
//
#include <hip/hip_runtime.h>
#include <hip/hip_bf16.h>
#include <math.h>

typedef __attribute__((ext_vector_type(16))) _Float16 v16h;
typedef __attribute__((ext_vector_type(8)))  float    v8f;

// ---------------- problem constants ----------------
constexpr int   BATCH   = 32;
constexpr int   CHN     = 3;
constexpr int   HW      = 512 * 512;          // 262144 = 1<<18
constexpr int   HW_LOG2 = 18;
constexpr int   PIX     = BATCH * HW;         // 8,388,608
constexpr float IO_C    = 240.0f;
constexpr float INV_IO  = 1.0f / 240.0f;
constexpr float BETA_C  = 0.15f;
constexpr float PI_C    = 3.14159265358979323846f;

// heavy-pass launch geometry: 1024*256 = 262144 threads -> exactly 32 iters/thread
constexpr int NBLK = 1024;
constexpr int NTH  = 256;

// ---------------- workspace word map (4-byte words) ----------------
constexpr int W_CNT      = 0;          // f32 masked count
constexpr int W_SUM      = 1;          // f32 x3
constexpr int W_GRAM     = 4;          // f32 x6 (00,01,02,11,12,22)
constexpr int W_E        = 10;         // f32 E[3][2] row-major (col0=mid eig, col1=largest)
constexpr int W_RMIN     = 16;         // u32 rank (0-based) for 1st percentile of phi
constexpr int W_RMAX     = 17;         // u32 rank for 99th percentile of phi
constexpr int W_SELMIN   = 18;         // f32 lo, f32 width ; u32 rem at +2
constexpr int W_SELMAX   = 21;
constexpr int W_CMIN0    = 24;         // u32 encoded-min/max of C rows
constexpr int W_CMAX0    = 25;
constexpr int W_CMIN1    = 26;
constexpr int W_CMAX1    = 27;
constexpr int W_M        = 28;         // f32 M[2][3]
constexpr int W_AF       = 37;         // f32 Afinal[3][3]
constexpr int W_HP       = 64;                 // coarse phi hist (8192)
constexpr int W_HFMIN    = W_HP + 8192;        // fine hist for minPhi (4096)
constexpr int W_HFMAX    = W_HFMIN + 4096;     // fine hist for maxPhi (4096)
constexpr int W_HC0      = W_HFMAX + 4096;     // hist of C row0 (4096)
constexpr int W_HC1      = W_HC0 + 4096;       // hist of C row1 (4096)
constexpr int W_TOTAL    = W_HC1 + 4096;       // ~98.6 KB

// ---------------- helpers ----------------
__device__ __forceinline__ void load_od(const float* __restrict__ x, int p,
                                        float& o0, float& o1, float& o2) {
    int b = p >> HW_LOG2;
    int s = p & (HW - 1);
    const float* xb = x + ((size_t)b * CHN << HW_LOG2) + s;
    o0 = -logf((xb[0]        + 1.0f) * INV_IO);
    o1 = -logf((xb[HW]       + 1.0f) * INV_IO);
    o2 = -logf((xb[2 * HW]   + 1.0f) * INV_IO);
}

__device__ __forceinline__ unsigned fenc(float f) {
    unsigned u = __float_as_uint(f);
    return (u & 0x80000000u) ? ~u : (u | 0x80000000u);
}
__device__ __forceinline__ float fdec(unsigned e) {
    unsigned u = (e & 0x80000000u) ? (e & 0x7FFFFFFFu) : ~e;
    return __uint_as_float(u);
}

// cooperative order-statistic selection from a histogram; call with 256 threads.
// out2[0]=bin, out2[1]=rank remainder inside bin (written by thread 0)
__device__ void hist_select(const unsigned* __restrict__ h, int nbins, unsigned rank,
                            unsigned* sh /*>=256*/, unsigned* out2) {
    __syncthreads();
    int t = threadIdx.x;
    int chunk = nbins >> 8;
    unsigned s = 0;
    for (int i = 0; i < chunk; ++i) s += h[t * chunk + i];
    sh[t] = s;
    __syncthreads();
    if (t == 0) {
        unsigned cum = 0;
        int ci = 0;
        for (; ci < 255; ++ci) {
            if (cum + sh[ci] > rank) break;
            cum += sh[ci];
        }
        int bin = ci * chunk + chunk - 1;
        for (int j = 0; j < chunk; ++j) {
            unsigned v = h[ci * chunk + j];
            if (cum + v > rank) { bin = ci * chunk + j; break; }
            cum += v;
        }
        out2[0] = (unsigned)bin;
        out2[1] = rank - cum;
    }
    __syncthreads();
}

// ---------------- kernels ----------------
__global__ void k_init(unsigned* wsu) {
    int i = blockIdx.x * blockDim.x + threadIdx.x;
    if (i < W_TOTAL) {
        unsigned v = 0;
        if (i == W_CMIN0 || i == W_CMIN1) v = 0xFFFFFFFFu;
        wsu[i] = v;
    }
}

// Pass 1: masked count, sum, Gram (second moments)
__global__ void k_moments(const float* __restrict__ x, float* wsf) {
    int tid = blockIdx.x * blockDim.x + threadIdx.x;
    int total = gridDim.x * blockDim.x;
    float acc[10];
    #pragma unroll
    for (int i = 0; i < 10; ++i) acc[i] = 0.0f;
    for (int p = tid; p < PIX; p += total) {
        float o0, o1, o2;
        load_od(x, p, o0, o1, o2);
        if (o0 >= BETA_C && o1 >= BETA_C && o2 >= BETA_C) {
            acc[0] += 1.0f;
            acc[1] += o0; acc[2] += o1; acc[3] += o2;
            acc[4] += o0 * o0; acc[5] += o0 * o1; acc[6] += o0 * o2;
            acc[7] += o1 * o1; acc[8] += o1 * o2; acc[9] += o2 * o2;
        }
    }
    #pragma unroll
    for (int i = 0; i < 10; ++i)
        for (int o = 16; o; o >>= 1) acc[i] += __shfl_xor(acc[i], o, 32);
    if ((threadIdx.x & 31) == 0) {
        atomicAdd(&wsf[W_CNT], acc[0]);
        #pragma unroll
        for (int i = 0; i < 3; ++i) atomicAdd(&wsf[W_SUM + i],  acc[1 + i]);
        #pragma unroll
        for (int i = 0; i < 6; ++i) atomicAdd(&wsf[W_GRAM + i], acc[4 + i]);
    }
}

// Pass 2 (1 thread): covariance -> Jacobi eigensolve -> E (two largest), ranks
__global__ void k_eigen(float* wsf, unsigned* wsu) {
    float n = wsf[W_CNT];
    if (n < 3.0f) n = 3.0f;
    float mu0 = wsf[W_SUM + 0] / n, mu1 = wsf[W_SUM + 1] / n, mu2 = wsf[W_SUM + 2] / n;
    float inm1 = 1.0f / (n - 1.0f);
    float A[3][3], V[3][3];
    A[0][0] = (wsf[W_GRAM + 0] - n * mu0 * mu0) * inm1;
    A[0][1] = A[1][0] = (wsf[W_GRAM + 1] - n * mu0 * mu1) * inm1;
    A[0][2] = A[2][0] = (wsf[W_GRAM + 2] - n * mu0 * mu2) * inm1;
    A[1][1] = (wsf[W_GRAM + 3] - n * mu1 * mu1) * inm1;
    A[1][2] = A[2][1] = (wsf[W_GRAM + 4] - n * mu1 * mu2) * inm1;
    A[2][2] = (wsf[W_GRAM + 5] - n * mu2 * mu2) * inm1;
    for (int i = 0; i < 3; ++i)
        for (int j = 0; j < 3; ++j) V[i][j] = (i == j) ? 1.0f : 0.0f;
    const int PQ[3][2] = {{0, 1}, {0, 2}, {1, 2}};
    for (int sweep = 0; sweep < 30; ++sweep) {
        float off = fabsf(A[0][1]) + fabsf(A[0][2]) + fabsf(A[1][2]);
        if (off < 1e-12f) break;
        for (int r = 0; r < 3; ++r) {
            int p = PQ[r][0], q = PQ[r][1];
            float apq = A[p][q];
            if (fabsf(apq) < 1e-20f) continue;
            float theta = (A[q][q] - A[p][p]) / (2.0f * apq);
            float t = ((theta >= 0.0f) ? 1.0f : -1.0f) /
                      (fabsf(theta) + sqrtf(theta * theta + 1.0f));
            float c = 1.0f / sqrtf(t * t + 1.0f);
            float s = t * c;
            for (int k = 0; k < 3; ++k) {
                float akp = A[k][p], akq = A[k][q];
                A[k][p] = c * akp - s * akq;
                A[k][q] = s * akp + c * akq;
            }
            for (int k = 0; k < 3; ++k) {
                float apk = A[p][k], aqk = A[q][k];
                A[p][k] = c * apk - s * aqk;
                A[q][k] = s * apk + c * aqk;
            }
            for (int k = 0; k < 3; ++k) {
                float vkp = V[k][p], vkq = V[k][q];
                V[k][p] = c * vkp - s * vkq;
                V[k][q] = s * vkp + c * vkq;
            }
        }
    }
    // sort eigenvalues ascending
    int idx[3] = {0, 1, 2};
    float ev[3] = {A[0][0], A[1][1], A[2][2]};
    for (int i = 0; i < 2; ++i)
        for (int j = 0; j < 2 - i; ++j)
            if (ev[idx[j]] > ev[idx[j + 1]]) { int tmp = idx[j]; idx[j] = idx[j + 1]; idx[j + 1] = tmp; }
    // E = columns [mid, largest], deterministic sign (largest-magnitude comp positive)
    for (int jj = 0; jj < 2; ++jj) {
        int c = idx[1 + jj];
        float e0 = V[0][c], e1 = V[1][c], e2 = V[2][c];
        float a0 = fabsf(e0), a1 = fabsf(e1), a2 = fabsf(e2);
        float m = (a0 >= a1 && a0 >= a2) ? e0 : ((a1 >= a2) ? e1 : e2);
        float sg = (m < 0.0f) ? -1.0f : 1.0f;
        wsf[W_E + 0 * 2 + jj] = e0 * sg;
        wsf[W_E + 1 * 2 + jj] = e1 * sg;
        wsf[W_E + 2 * 2 + jj] = e2 * sg;
    }
    // percentile ranks (0-based) into the masked-phi distribution
    int ni = (int)(wsf[W_CNT] + 0.5f);
    if (ni < 2) ni = 2;
    wsu[W_RMIN] = (unsigned)((int)(1.0f + floorf(0.01f * 1.0f  * (float)(ni - 1) + 0.5f)) - 1);
    wsu[W_RMAX] = (unsigned)((int)(1.0f + floorf(0.01f * 99.0f * (float)(ni - 1) + 0.5f)) - 1);
}

// Pass 3: coarse phi histogram (8192 bins over [-pi, pi]), LDS then global flush
__global__ void k_phi_hist(const float* __restrict__ x, const float* wsf, unsigned* wsu) {
    __shared__ unsigned sh[8192];
    for (int i = threadIdx.x; i < 8192; i += blockDim.x) sh[i] = 0;
    __syncthreads();
    float e00 = wsf[W_E + 0], e01 = wsf[W_E + 1];
    float e10 = wsf[W_E + 2], e11 = wsf[W_E + 3];
    float e20 = wsf[W_E + 4], e21 = wsf[W_E + 5];
    int tid = blockIdx.x * blockDim.x + threadIdx.x;
    int total = gridDim.x * blockDim.x;
    const float scale = 8192.0f / (2.0f * PI_C);
    for (int p = tid; p < PIX; p += total) {
        float o0, o1, o2;
        load_od(x, p, o0, o1, o2);
        if (o0 >= BETA_C && o1 >= BETA_C && o2 >= BETA_C) {
            float t0 = o0 * e00 + o1 * e10 + o2 * e20;
            float t1 = o0 * e01 + o1 * e11 + o2 * e21;
            float phi = atan2f(t1, t0);
            int bin = (int)((phi + PI_C) * scale);
            bin = min(max(bin, 0), 8191);
            atomicAdd(&sh[bin], 1u);
        }
    }
    __syncthreads();
    for (int i = threadIdx.x; i < 8192; i += blockDim.x)
        if (sh[i]) atomicAdd(&wsu[W_HP + i], sh[i]);
}

// Pass 4 (1 block): locate coarse bins for both percentiles
__global__ void k_phi_select(float* wsf, unsigned* wsu) {
    __shared__ unsigned sh[256];
    __shared__ unsigned res[4];
    const float W = (2.0f * PI_C) / 8192.0f;
    hist_select(&wsu[W_HP], 8192, wsu[W_RMIN], sh, &res[0]);
    hist_select(&wsu[W_HP], 8192, wsu[W_RMAX], sh, &res[2]);
    if (threadIdx.x == 0) {
        wsf[W_SELMIN + 0] = -PI_C + (float)res[0] * W;
        wsf[W_SELMIN + 1] = W;
        wsu[W_SELMIN + 2] = res[1];
        wsf[W_SELMAX + 0] = -PI_C + (float)res[2] * W;
        wsf[W_SELMAX + 1] = W;
        wsu[W_SELMAX + 2] = res[3];
    }
}

// Pass 5: fine phi histograms inside the two selected coarse bins
__global__ void k_phi_fine(const float* __restrict__ x, const float* wsf, unsigned* wsu) {
    __shared__ unsigned shmin[4096];
    __shared__ unsigned shmax[4096];
    for (int i = threadIdx.x; i < 4096; i += blockDim.x) { shmin[i] = 0; shmax[i] = 0; }
    __syncthreads();
    float e00 = wsf[W_E + 0], e01 = wsf[W_E + 1];
    float e10 = wsf[W_E + 2], e11 = wsf[W_E + 3];
    float e20 = wsf[W_E + 4], e21 = wsf[W_E + 5];
    float loA = wsf[W_SELMIN + 0], wA = wsf[W_SELMIN + 1];
    float loB = wsf[W_SELMAX + 0], wB = wsf[W_SELMAX + 1];
    float sA = 4096.0f / wA, sB = 4096.0f / wB;
    int tid = blockIdx.x * blockDim.x + threadIdx.x;
    int total = gridDim.x * blockDim.x;
    for (int p = tid; p < PIX; p += total) {
        float o0, o1, o2;
        load_od(x, p, o0, o1, o2);
        if (o0 >= BETA_C && o1 >= BETA_C && o2 >= BETA_C) {
            float t0 = o0 * e00 + o1 * e10 + o2 * e20;
            float t1 = o0 * e01 + o1 * e11 + o2 * e21;
            float phi = atan2f(t1, t0);
            int ba = (int)((phi - loA) * sA);
            if (ba >= 0 && ba < 4096) atomicAdd(&shmin[ba], 1u);
            int bb = (int)((phi - loB) * sB);
            if (bb >= 0 && bb < 4096) atomicAdd(&shmax[bb], 1u);
        }
    }
    __syncthreads();
    for (int i = threadIdx.x; i < 4096; i += blockDim.x) {
        if (shmin[i]) atomicAdd(&wsu[W_HFMIN + i], shmin[i]);
        if (shmax[i]) atomicAdd(&wsu[W_HFMAX + i], shmax[i]);
    }
}

// Pass 6 (1 block): finalize minPhi/maxPhi, build HE and M = inv(HE^T HE) HE^T
__global__ void k_phi_final(float* wsf, unsigned* wsu) {
    __shared__ unsigned sh[256];
    __shared__ unsigned res[4];
    hist_select(&wsu[W_HFMIN], 4096, wsu[W_SELMIN + 2], sh, &res[0]);
    hist_select(&wsu[W_HFMAX], 4096, wsu[W_SELMAX + 2], sh, &res[2]);
    if (threadIdx.x == 0) {
        float minPhi = wsf[W_SELMIN + 0] + ((float)res[0] + 0.5f) * (wsf[W_SELMIN + 1] * (1.0f / 4096.0f));
        float maxPhi = wsf[W_SELMAX + 0] + ((float)res[2] + 0.5f) * (wsf[W_SELMAX + 1] * (1.0f / 4096.0f));
        float E[3][2];
        for (int i = 0; i < 3; ++i) { E[i][0] = wsf[W_E + i * 2]; E[i][1] = wsf[W_E + i * 2 + 1]; }
        float cmn = cosf(minPhi), smn = sinf(minPhi);
        float cmx = cosf(maxPhi), smx = sinf(maxPhi);
        float vMin[3], vMax[3];
        for (int i = 0; i < 3; ++i) {
            vMin[i] = E[i][0] * cmn + E[i][1] * smn;
            vMax[i] = E[i][0] * cmx + E[i][1] * smx;
        }
        float h0[3], h1[3];
        if (vMin[0] > vMax[0]) {
            for (int i = 0; i < 3; ++i) { h0[i] = vMin[i]; h1[i] = vMax[i]; }
        } else {
            for (int i = 0; i < 3; ++i) { h0[i] = vMax[i]; h1[i] = vMin[i]; }
        }
        float a = h0[0]*h0[0] + h0[1]*h0[1] + h0[2]*h0[2];
        float b = h0[0]*h1[0] + h0[1]*h1[1] + h0[2]*h1[2];
        float c = h1[0]*h1[0] + h1[1]*h1[1] + h1[2]*h1[2];
        float det = a * c - b * b;
        if (fabsf(det) < 1e-20f) det = 1e-20f;
        float id = 1.0f / det;
        for (int j = 0; j < 3; ++j) {
            wsf[W_M + 0 * 3 + j] = ( c * h0[j] - b * h1[j]) * id;
            wsf[W_M + 1 * 3 + j] = (-b * h0[j] + a * h1[j]) * id;
        }
    }
}

// Pass 7: min/max of C = M @ OD over ALL pixels (ordered-float encoded atomics)
__global__ void k_cminmax(const float* __restrict__ x, const float* wsf, unsigned* wsu) {
    float m00 = wsf[W_M + 0], m01 = wsf[W_M + 1], m02 = wsf[W_M + 2];
    float m10 = wsf[W_M + 3], m11 = wsf[W_M + 4], m12 = wsf[W_M + 5];
    int tid = blockIdx.x * blockDim.x + threadIdx.x;
    int total = gridDim.x * blockDim.x;
    unsigned mn0 = 0xFFFFFFFFu, mx0 = 0u, mn1 = 0xFFFFFFFFu, mx1 = 0u;
    for (int p = tid; p < PIX; p += total) {
        float o0, o1, o2;
        load_od(x, p, o0, o1, o2);
        unsigned e0 = fenc(m00 * o0 + m01 * o1 + m02 * o2);
        unsigned e1 = fenc(m10 * o0 + m11 * o1 + m12 * o2);
        mn0 = min(mn0, e0); mx0 = max(mx0, e0);
        mn1 = min(mn1, e1); mx1 = max(mx1, e1);
    }
    for (int o = 16; o; o >>= 1) {
        mn0 = min(mn0, (unsigned)__shfl_xor(mn0, o, 32));
        mx0 = max(mx0, (unsigned)__shfl_xor(mx0, o, 32));
        mn1 = min(mn1, (unsigned)__shfl_xor(mn1, o, 32));
        mx1 = max(mx1, (unsigned)__shfl_xor(mx1, o, 32));
    }
    if ((threadIdx.x & 31) == 0) {
        atomicMin(&wsu[W_CMIN0], mn0);
        atomicMax(&wsu[W_CMAX0], mx0);
        atomicMin(&wsu[W_CMIN1], mn1);
        atomicMax(&wsu[W_CMAX1], mx1);
    }
}

// Pass 8: histograms of C row0 and row1 over their [min,max] ranges
__global__ void k_chist(const float* __restrict__ x, const float* wsf, unsigned* wsu) {
    __shared__ unsigned sh0[4096];
    __shared__ unsigned sh1[4096];
    for (int i = threadIdx.x; i < 4096; i += blockDim.x) { sh0[i] = 0; sh1[i] = 0; }
    __syncthreads();
    float m00 = wsf[W_M + 0], m01 = wsf[W_M + 1], m02 = wsf[W_M + 2];
    float m10 = wsf[W_M + 3], m11 = wsf[W_M + 4], m12 = wsf[W_M + 5];
    float lo0 = fdec(wsu[W_CMIN0]), hi0 = fdec(wsu[W_CMAX0]);
    float lo1 = fdec(wsu[W_CMIN1]), hi1 = fdec(wsu[W_CMAX1]);
    float w0 = hi0 - lo0; if (w0 <= 0.0f) w0 = 1e-6f;
    float w1 = hi1 - lo1; if (w1 <= 0.0f) w1 = 1e-6f;
    float s0 = 4096.0f / w0, s1 = 4096.0f / w1;
    int tid = blockIdx.x * blockDim.x + threadIdx.x;
    int total = gridDim.x * blockDim.x;
    for (int p = tid; p < PIX; p += total) {
        float o0, o1, o2;
        load_od(x, p, o0, o1, o2);
        float c0 = m00 * o0 + m01 * o1 + m02 * o2;
        float c1 = m10 * o0 + m11 * o1 + m12 * o2;
        int b0 = min(max((int)((c0 - lo0) * s0), 0), 4095);
        int b1 = min(max((int)((c1 - lo1) * s1), 0), 4095);
        atomicAdd(&sh0[b0], 1u);
        atomicAdd(&sh1[b1], 1u);
    }
    __syncthreads();
    for (int i = threadIdx.x; i < 4096; i += blockDim.x) {
        if (sh0[i]) atomicAdd(&wsu[W_HC0 + i], sh0[i]);
        if (sh1[i]) atomicAdd(&wsu[W_HC1 + i], sh1[i]);
    }
}

// Pass 9 (1 block): 99th percentiles of C rows -> fused 3x3 matrix Afinal
__global__ void k_cfinal(float* wsf, unsigned* wsu) {
    __shared__ unsigned sh[256];
    __shared__ unsigned res[4];
    unsigned rc = (unsigned)((int)(1.0f + floorf(0.99f * (float)(PIX - 1) + 0.5f)) - 1);
    hist_select(&wsu[W_HC0], 4096, rc, sh, &res[0]);
    hist_select(&wsu[W_HC1], 4096, rc, sh, &res[2]);
    if (threadIdx.x == 0) {
        float lo0 = fdec(wsu[W_CMIN0]), hi0 = fdec(wsu[W_CMAX0]);
        float lo1 = fdec(wsu[W_CMIN1]), hi1 = fdec(wsu[W_CMAX1]);
        float w0 = hi0 - lo0; if (w0 <= 0.0f) w0 = 1e-6f;
        float w1 = hi1 - lo1; if (w1 <= 0.0f) w1 = 1e-6f;
        float maxC0 = lo0 + ((float)res[0] + 0.5f) * (w0 * (1.0f / 4096.0f));
        float maxC1 = lo1 + ((float)res[2] + 0.5f) * (w1 * (1.0f / 4096.0f));
        if (fabsf(maxC0) < 1e-12f) maxC0 = 1e-12f;
        if (fabsf(maxC1) < 1e-12f) maxC1 = 1e-12f;
        float s0 = 1.9705f / maxC0;
        float s1 = 1.0308f / maxC1;
        const float HR[3][2] = {{0.5626f, 0.2159f}, {0.7201f, 0.8012f}, {0.4062f, 0.5581f}};
        for (int r = 0; r < 3; ++r)
            for (int k = 0; k < 3; ++k)
                wsf[W_AF + r * 3 + k] = HR[r][0] * s0 * wsf[W_M + 0 * 3 + k]
                                      + HR[r][1] * s1 * wsf[W_M + 1 * 3 + k];
    }
}

// Pass 10: recolor. out(3xN) = min(240*exp(Afinal @ log p), 255) via v_wmma_f32_16x16x32_f16.
// A: rows 0..2 = fused 3x3 (lanes 0..2, K=0..2). B: column n = 16 pixels' logs.
// Two WMMAs per wave-iteration cover 32 pixels; shfl_xor(16) keeps all loads/stores
// fully coalesced across 32 lanes. PIX % (NBLK*NTH) == 0 -> full waves, EXEC all-ones.
__global__ void k_out_wmma(const float* __restrict__ x, const float* wsf,
                           float* __restrict__ out) {
    int tid = blockIdx.x * blockDim.x + threadIdx.x;
    int total = gridDim.x * blockDim.x;
    unsigned lane = threadIdx.x & 31u;
    bool lo = lane < 16u;

    // loop-invariant A matrix in WMMA register layout
    float a0 = 0.0f, a1 = 0.0f, a2 = 0.0f;
    if (lane < 3u) {
        a0 = wsf[W_AF + lane * 3 + 0];
        a1 = wsf[W_AF + lane * 3 + 1];
        a2 = wsf[W_AF + lane * 3 + 2];
    }
    v16h A = {};
    A[0] = (_Float16)a0; A[1] = (_Float16)a1; A[2] = (_Float16)a2;

    for (int p = tid; p < PIX; p += total) {
        int b = p >> HW_LOG2;
        int s = p & (HW - 1);
        const float* xb = x + ((size_t)b * CHN << HW_LOG2) + s;
        float l0 = logf((xb[0]      + 1.0f) * INV_IO);
        float l1 = logf((xb[HW]     + 1.0f) * INV_IO);
        float l2 = logf((xb[2 * HW] + 1.0f) * INV_IO);
        // logs of the upper half-wave's pixels, delivered to lanes 0..15
        float m0 = __shfl_xor(l0, 16, 32);
        float m1 = __shfl_xor(l1, 16, 32);
        float m2 = __shfl_xor(l2, 16, 32);

        v16h B0 = {}, B1 = {};
        B0[0] = (_Float16)(lo ? l0 : 0.0f);
        B0[1] = (_Float16)(lo ? l1 : 0.0f);
        B0[2] = (_Float16)(lo ? l2 : 0.0f);
        B1[0] = (_Float16)(lo ? m0 : 0.0f);
        B1[1] = (_Float16)(lo ? m1 : 0.0f);
        B1[2] = (_Float16)(lo ? m2 : 0.0f);

        v8f C0 = {}, C1 = {};
        C0 = __builtin_amdgcn_wmma_f32_16x16x32_f16(false, A, false, B0,
                                                    (short)0, C0, false, false);
        C1 = __builtin_amdgcn_wmma_f32_16x16x32_f16(false, A, false, B1,
                                                    (short)0, C1, false, false);

        float* ob = out + ((size_t)b * CHN << HW_LOG2) + s;
        #pragma unroll
        for (int c = 0; c < 3; ++c) {
            float g1 = __shfl_xor(C1[c], 16, 32);  // group1 result to lanes 16..31
            float v  = lo ? C0[c] : g1;
            ob[c * HW] = fminf(IO_C * expf(v), 255.0f);
        }
    }
}

// ---------------- launch ----------------
extern "C" void kernel_launch(void* const* d_in, const int* in_sizes, int n_in,
                              void* d_out, int out_size, void* d_ws, size_t ws_size,
                              hipStream_t stream) {
    (void)in_sizes; (void)n_in; (void)out_size; (void)ws_size;
    const float* x   = (const float*)d_in[0];
    float*       out = (float*)d_out;
    float*       wsf = (float*)d_ws;
    unsigned*    wsu = (unsigned*)d_ws;

    k_init      <<<(W_TOTAL + 255) / 256, 256, 0, stream>>>(wsu);
    k_moments   <<<NBLK, NTH, 0, stream>>>(x, wsf);
    k_eigen     <<<1, 1,    0, stream>>>(wsf, wsu);
    k_phi_hist  <<<NBLK, NTH, 0, stream>>>(x, wsf, wsu);
    k_phi_select<<<1, 256,  0, stream>>>(wsf, wsu);
    k_phi_fine  <<<NBLK, NTH, 0, stream>>>(x, wsf, wsu);
    k_phi_final <<<1, 256,  0, stream>>>(wsf, wsu);
    k_cminmax   <<<NBLK, NTH, 0, stream>>>(x, wsf, wsu);
    k_chist     <<<NBLK, NTH, 0, stream>>>(x, wsf, wsu);
    k_cfinal    <<<1, 256,  0, stream>>>(wsf, wsu);
    k_out_wmma  <<<NBLK, NTH, 0, stream>>>(x, wsf, out);
}
